// CounterfactualMamba2_64115271795225
// MI455X (gfx1250) — matrix operand order repaired and checked by reading, not compiled, not run on hardware
//
#include <hip/hip_runtime.h>
#include <hip/hip_bf16.h>
#include <math.h>

typedef __bf16 bf16_t;
typedef bf16_t v16bf __attribute__((ext_vector_type(16)));
typedef bf16_t v8bf  __attribute__((ext_vector_type(8)));
typedef float  v8f   __attribute__((ext_vector_type(8)));

#define D_MODEL   512
#define D_STATE   64
#define HEADDIM   64
#define D_INNER   1024
#define NHEADS    16
#define D_IN_PROJ 2192
#define XBC_DIM   1152
#define BATCH     4
#define LQ        1024
#define LTOT      2048
#define NTOK      (BATCH * LTOT)   // 8192 tokens per branch
#define NTOKQ     (BATCH * LQ)     // 4096 query tokens

// ---------------------------------------------------------------------------
// WMMA fragment loaders per CDNA5 ISA wave32 layouts (CDNA5 ISA)
// ---------------------------------------------------------------------------

// A-matrix 16x32 bf16: lanes0-15 hold M=lane, K = {0..7, 16..23};
// lanes16-31 hold M=lane-16, K = {8..15, 24..31}.
__device__ __forceinline__ v16bf load_a_frag(const bf16_t* A, int lda, int m0,
                                             int k0, int lane) {
  int mr = lane & 15;
  int kb = (lane >> 4) << 3;  // 0 or 8
  const bf16_t* p = A + (size_t)(m0 + mr) * lda + (k0 + kb);
  v8bf lo = *(const v8bf*)(p);
  v8bf hi = *(const v8bf*)(p + 16);
  v16bf r;
#pragma unroll
  for (int i = 0; i < 8; ++i) { r[i] = lo[i]; r[8 + i] = hi[i]; }
  return r;
}

// B-matrix 32x16 bf16 (from transposed NxK weight copy): lane holds column
// N = lane&15; lanes0-15 hold K = {0..7, 16..23}, lanes16-31 hold K = {8..15, 24..31}.
__device__ __forceinline__ v16bf load_b_frag(const bf16_t* BT, int ldb, int n0,
                                             int k0, int lane) {
  int nc = lane & 15;
  int kb = (lane >> 4) << 3;  // 0 or 8
  const bf16_t* p = BT + (size_t)(n0 + nc) * ldb + (k0 + kb);
  v8bf lo = *(const v8bf*)(p);
  v8bf hi = *(const v8bf*)(p + 16);
  v16bf r;
#pragma unroll
  for (int i = 0; i < 8; ++i) { r[i] = lo[i]; r[8 + i] = hi[i]; }
  return r;
}

__device__ __forceinline__ v8f wmma_bf16(v16bf a, v16bf b, v8f c) {
  return __builtin_amdgcn_wmma_f32_16x16x32_bf16(false, a, false, b, (short)0,
                                                 c, false, false);
}

// Store one 16x16 f32 accumulator per ISA C/D layout:
// lanes0-15: VGPR r -> M=r; lanes16-31: VGPR r -> M=r+8; N=lane&15.
__device__ __forceinline__ void store_acc(float* C, int N, int m0, int n0,
                                          int lane, v8f acc, float bv) {
  int nc = n0 + (lane & 15);
  int mr = m0 + ((lane >> 4) << 3);
#pragma unroll
  for (int r = 0; r < 8; ++r) C[(size_t)(mr + r) * N + nc] = acc[r] + bv;
}

// ---------------------------------------------------------------------------
// GEMM: C[MxN] = A[MxK](bf16) * B[KxN] (given as BT[NxK] bf16) + bias (opt)
// Block = 256 threads = 8 waves arranged 2(M) x 4(N).
// Wave tile = 32x32 (4 independent accumulators -> hazard-free WMMA interleave,
// half the fragment traffic per FLOP). Block tile = 64x128.
// M must be a multiple of 64; ragged N handled wave-uniformly.
// ---------------------------------------------------------------------------
__global__ void __launch_bounds__(256)
gemm_bf16f32(const bf16_t* __restrict__ A, const bf16_t* __restrict__ BT,
             float* __restrict__ C, int M, int N, int K,
             const float* __restrict__ bias) {
  int lane = threadIdx.x & 31;
  int wave = threadIdx.x >> 5;
  int m0 = blockIdx.y * 64 + (wave & 1) * 32;
  int n0 = blockIdx.x * 128 + (wave >> 1) * 32;
  if (m0 >= M || n0 >= N) return;       // wave-uniform exit
  bool n1ok = (n0 + 16) < N;            // wave-uniform

  v8f acc00 = {}, acc01 = {}, acc10 = {}, acc11 = {};
  if (n1ok) {
    for (int k0 = 0; k0 < K; k0 += 32) {
      v16bf a0 = load_a_frag(A, K, m0, k0, lane);
      v16bf a1 = load_a_frag(A, K, m0 + 16, k0, lane);
      v16bf b0 = load_b_frag(BT, K, n0, k0, lane);
      v16bf b1 = load_b_frag(BT, K, n0 + 16, k0, lane);
      acc00 = wmma_bf16(a0, b0, acc00);
      acc10 = wmma_bf16(a1, b0, acc10);
      acc01 = wmma_bf16(a0, b1, acc01);
      acc11 = wmma_bf16(a1, b1, acc11);
    }
  } else {
    for (int k0 = 0; k0 < K; k0 += 32) {
      v16bf a0 = load_a_frag(A, K, m0, k0, lane);
      v16bf a1 = load_a_frag(A, K, m0 + 16, k0, lane);
      v16bf b0 = load_b_frag(BT, K, n0, k0, lane);
      acc00 = wmma_bf16(a0, b0, acc00);
      acc10 = wmma_bf16(a1, b0, acc10);
    }
  }

  float bv0 = bias ? bias[n0 + (lane & 15)] : 0.0f;
  store_acc(C, N, m0, n0, lane, acc00, bv0);
  store_acc(C, N, m0 + 16, n0, lane, acc10, bv0);
  if (n1ok) {
    float bv1 = bias ? bias[n0 + 16 + (lane & 15)] : 0.0f;
    store_acc(C, N, m0, n0 + 16, lane, acc01, bv1);
    store_acc(C, N, m0 + 16, n0 + 16, lane, acc11, bv1);
  }
}

// ---------------------------------------------------------------------------
// Weight convert + transpose: W[K x N] fp32 -> WT[N x K] bf16
// ---------------------------------------------------------------------------
__global__ void convert_transpose(const float* __restrict__ W,
                                  bf16_t* __restrict__ WT, int K, int N) {
  size_t idx = (size_t)blockIdx.x * blockDim.x + threadIdx.x;
  if (idx >= (size_t)K * N) return;
  int k = (int)(idx % K);
  int n = (int)(idx / K);
  WT[(size_t)n * K + k] = (bf16_t)W[(size_t)k * N + n];
}

// ---------------------------------------------------------------------------
// Build rmsnorm'd branch input (concat kv|flip(kv) with q) -> bf16 (NTOK x 512)
// One block (256 threads) per token.
// ---------------------------------------------------------------------------
__global__ void __launch_bounds__(256)
build_norm_input(const float* __restrict__ q, const float* __restrict__ kv,
                 const float* __restrict__ w, bf16_t* __restrict__ out,
                 int flip) {
  __shared__ float red[256];
  int tok = blockIdx.x;            // 0..NTOK-1
  int b = tok >> 11;               // /2048
  int l = tok & 2047;
  const float* src;
  if (l < LQ) {
    int lk = flip ? (LQ - 1 - l) : l;
    src = kv + ((size_t)b * LQ + lk) * D_MODEL;
  } else {
    src = q + ((size_t)b * LQ + (l - LQ)) * D_MODEL;
  }
  float v[2];
  float ss = 0.0f;
#pragma unroll
  for (int j = 0; j < 2; ++j) {
    int i = threadIdx.x + 256 * j;
    v[j] = src[i];
    ss += v[j] * v[j];
  }
  red[threadIdx.x] = ss;
  __syncthreads();
  for (int s = 128; s > 0; s >>= 1) {
    if (threadIdx.x < s) red[threadIdx.x] += red[threadIdx.x + s];
    __syncthreads();
  }
  float scale = rsqrtf(red[0] * (1.0f / D_MODEL) + 1e-5f);
#pragma unroll
  for (int j = 0; j < 2; ++j) {
    int i = threadIdx.x + 256 * j;
    out[(size_t)tok * D_MODEL + i] = (bf16_t)(v[j] * scale * w[i]);
  }
}

// ---------------------------------------------------------------------------
// Causal depthwise conv (taps=4) + SiLU over xBC channels of zxbcdt
// ---------------------------------------------------------------------------
__global__ void conv_silu_kernel(const float* __restrict__ zx,
                                 const float* __restrict__ cw,
                                 const float* __restrict__ cb,
                                 float* __restrict__ xbc) {
  size_t idx = (size_t)blockIdx.x * blockDim.x + threadIdx.x;
  if (idx >= (size_t)NTOK * XBC_DIM) return;
  int c = (int)(idx % XBC_DIM);
  size_t tok = idx / XBC_DIM;
  int l = (int)(tok & 2047);
  size_t brow = tok - l;
  float acc = cb[c];
#pragma unroll
  for (int t = 0; t < 4; ++t) {
    int ls = l - 3 + t;
    if (ls >= 0)
      acc += cw[c * 4 + t] * zx[(brow + ls) * D_IN_PROJ + D_INNER + c];
  }
  xbc[idx] = acc / (1.0f + expf(-acc));  // silu
}

// ---------------------------------------------------------------------------
// dt = softplus(dt_raw + bias); dA = exp(-exp(A_log)*dt)
// ---------------------------------------------------------------------------
__global__ void dt_kernel(const float* __restrict__ zx,
                          const float* __restrict__ dt_bias,
                          const float* __restrict__ A_log,
                          float* __restrict__ dtv, float* __restrict__ dAv) {
  int idx = blockIdx.x * blockDim.x + threadIdx.x;
  if (idx >= NTOK * NHEADS) return;
  int h = idx & 15;
  size_t tok = (size_t)(idx >> 4);
  float x = zx[tok * D_IN_PROJ + 2 * D_INNER + 2 * D_STATE + h] + dt_bias[h];
  float dt = (x > 20.0f) ? x : log1pf(expf(x));
  dtv[idx] = dt;
  dAv[idx] = expf(-expf(A_log[h]) * dt);
}

// ---------------------------------------------------------------------------
// SSD scan: block per (batch, head), 64 threads (one per HEADDIM row),
// 64-wide state in VGPRs, B/C broadcast through double-buffered LDS.
// y = scan(...) + D[h]*x fused.
// ---------------------------------------------------------------------------
__global__ void __launch_bounds__(64)
ssd_scan_kernel(const float* __restrict__ xbc, const float* __restrict__ dtv,
                const float* __restrict__ dAv, const float* __restrict__ Dp,
                float* __restrict__ y) {
  int bh = blockIdx.x;
  int b = bh >> 4;
  int h = bh & 15;
  int p = threadIdx.x;  // 0..63
  float hs[D_STATE];
#pragma unroll
  for (int n = 0; n < D_STATE; ++n) hs[n] = 0.0f;

  __shared__ float Bs[2][D_STATE];
  __shared__ float Cs[2][D_STATE];

  const float* xb = xbc + (size_t)b * LTOT * XBC_DIM;
  float dcoef = Dp[h];

  for (int t = 0; t < LTOT; ++t) {
    int buf = t & 1;
    Bs[buf][p] = xb[(size_t)t * XBC_DIM + D_INNER + p];
    Cs[buf][p] = xb[(size_t)t * XBC_DIM + D_INNER + D_STATE + p];
    __syncthreads();
    size_t ti = (size_t)b * LTOT + t;
    float dt_t = dtv[ti * NHEADS + h];
    float dA_t = dAv[ti * NHEADS + h];
    float x_t = xb[(size_t)t * XBC_DIM + h * HEADDIM + p];
    float a = dt_t * x_t;
    float acc = 0.0f;
#pragma unroll
    for (int n = 0; n < D_STATE; ++n) {
      hs[n] = hs[n] * dA_t + a * Bs[buf][n];
      acc += hs[n] * Cs[buf][n];
    }
    y[ti * D_INNER + h * HEADDIM + p] = acc + dcoef * x_t;
  }
}

// ---------------------------------------------------------------------------
// yz = y * silu(z); rmsnorm over D_INNER; output bf16 for out-proj GEMM
// ---------------------------------------------------------------------------
__global__ void __launch_bounds__(256)
gate_rmsnorm_kernel(const float* __restrict__ y, const float* __restrict__ zx,
                    const float* __restrict__ normw, bf16_t* __restrict__ out) {
  __shared__ float red[256];
  int tok = blockIdx.x;
  const float* z = zx + (size_t)tok * D_IN_PROJ;
  const float* yy = y + (size_t)tok * D_INNER;
  float v[4];
  float ss = 0.0f;
#pragma unroll
  for (int j = 0; j < 4; ++j) {
    int i = threadIdx.x + 256 * j;
    float zi = z[i];
    float t = yy[i] * (zi / (1.0f + expf(-zi)));
    v[j] = t;
    ss += t * t;
  }
  red[threadIdx.x] = ss;
  __syncthreads();
  for (int s = 128; s > 0; s >>= 1) {
    if (threadIdx.x < s) red[threadIdx.x] += red[threadIdx.x + s];
    __syncthreads();
  }
  float scale = rsqrtf(red[0] * (1.0f / D_INNER) + 1e-5f);
#pragma unroll
  for (int j = 0; j < 4; ++j) {
    int i = threadIdx.x + 256 * j;
    out[(size_t)tok * D_INNER + i] = (bf16_t)(v[j] * scale * normw[i]);
  }
}

// Slice query half of (B, LTOT, 512) -> (NTOKQ, 512) as bf16 (GEMM input)
__global__ void slice_q_to_bf16(const float* __restrict__ fs,
                                bf16_t* __restrict__ out) {
  size_t idx = (size_t)blockIdx.x * blockDim.x + threadIdx.x;
  if (idx >= (size_t)NTOKQ * D_MODEL) return;
  int col = (int)(idx & 511);
  size_t row = idx >> 9;          // 0..4095
  size_t b = row >> 10;
  size_t l = row & 1023;
  out[idx] = (bf16_t)fs[((b * LTOT) + LQ + l) * D_MODEL + col];
}

// Slice query half of (B, LTOT, 512) -> fp32 output (counterfactual branch)
__global__ void slice_q_copy(const float* __restrict__ cs,
                             float* __restrict__ out) {
  size_t idx = (size_t)blockIdx.x * blockDim.x + threadIdx.x;
  if (idx >= (size_t)NTOKQ * D_MODEL) return;
  int col = (int)(idx & 511);
  size_t row = idx >> 9;
  size_t b = row >> 10;
  size_t l = row & 1023;
  out[idx] = cs[((b * LTOT) + LQ + l) * D_MODEL + col];
}

// ---------------------------------------------------------------------------
// Host orchestration
// ---------------------------------------------------------------------------
extern "C" void kernel_launch(void* const* d_in, const int* in_sizes, int n_in,
                              void* d_out, int out_size, void* d_ws,
                              size_t ws_size, hipStream_t stream) {
  const float* q       = (const float*)d_in[0];
  const float* kv      = (const float*)d_in[1];
  const float* fnorm_w = (const float*)d_in[2];
  const float* cnorm_w = (const float*)d_in[3];
  const float* outp_w  = (const float*)d_in[20];
  const float* outp_b  = (const float*)d_in[21];

  // per-branch parameter bundles (f: idx 4..11, c: idx 12..19)
  struct Params {
    const float *Win, *convw, *convb, *dtbias, *Alog, *Dp, *normw, *Wout;
  };
  Params P[2];
  for (int br = 0; br < 2; ++br) {
    int o = 4 + 8 * br;
    P[br].Win    = (const float*)d_in[o + 0];
    P[br].convw  = (const float*)d_in[o + 1];
    P[br].convb  = (const float*)d_in[o + 2];
    P[br].dtbias = (const float*)d_in[o + 3];
    P[br].Alog   = (const float*)d_in[o + 4];
    P[br].Dp     = (const float*)d_in[o + 5];
    P[br].normw  = (const float*)d_in[o + 6];
    P[br].Wout   = (const float*)d_in[o + 7];
  }

  // workspace carve-up (reused by both branches, run sequentially)
  size_t off = 0;
  auto carve = [&](size_t bytes) {
    void* p = (char*)d_ws + off;
    off = (off + bytes + 255) & ~(size_t)255;
    return p;
  };
  bf16_t* WinT  = (bf16_t*)carve((size_t)D_IN_PROJ * D_MODEL * sizeof(bf16_t));
  bf16_t* WoutT = (bf16_t*)carve((size_t)D_MODEL * D_INNER * sizeof(bf16_t));
  bf16_t* outpT = (bf16_t*)carve((size_t)D_MODEL * D_MODEL * sizeof(bf16_t));
  bf16_t* Xn    = (bf16_t*)carve((size_t)NTOK * D_MODEL * sizeof(bf16_t));
  float*  zx    = (float*)carve((size_t)NTOK * D_IN_PROJ * sizeof(float));
  float*  xbc   = (float*)carve((size_t)NTOK * XBC_DIM * sizeof(float));
  float*  dtv   = (float*)carve((size_t)NTOK * NHEADS * sizeof(float));
  float*  dAv   = (float*)carve((size_t)NTOK * NHEADS * sizeof(float));
  float*  yb    = (float*)carve((size_t)NTOK * D_INNER * sizeof(float));
  bf16_t* ynorm = (bf16_t*)carve((size_t)NTOK * D_INNER * sizeof(bf16_t));
  float*  fs    = (float*)carve((size_t)NTOK * D_MODEL * sizeof(float));
  bf16_t* A3    = (bf16_t*)carve((size_t)NTOKQ * D_MODEL * sizeof(bf16_t));
  (void)ws_size;

  float* out_fact = (float*)d_out;                               // (4,1024,512)
  float* out_cf   = (float*)d_out + (size_t)NTOKQ * D_MODEL;     // (4,1024,512)

  // final projection weight transpose (shared, once)
  {
    size_t n = (size_t)D_MODEL * D_MODEL;
    convert_transpose<<<(unsigned)((n + 255) / 256), 256, 0, stream>>>(
        outp_w, outpT, D_MODEL, D_MODEL);
  }

  for (int br = 0; br < 2; ++br) {
    const Params& pp = P[br];
    const float* nw = (br == 0) ? fnorm_w : cnorm_w;

    // weight transposes -> bf16
    {
      size_t n = (size_t)D_MODEL * D_IN_PROJ;
      convert_transpose<<<(unsigned)((n + 255) / 256), 256, 0, stream>>>(
          pp.Win, WinT, D_MODEL, D_IN_PROJ);
      size_t m = (size_t)D_INNER * D_MODEL;
      convert_transpose<<<(unsigned)((m + 255) / 256), 256, 0, stream>>>(
          pp.Wout, WoutT, D_INNER, D_MODEL);
    }

    // rmsnorm'd branch input (flip kv for counterfactual branch)
    build_norm_input<<<NTOK, 256, 0, stream>>>(q, kv, nw, Xn, br);

    // in_proj GEMM: (8192 x 512) x (512 x 2192)
    {
      dim3 grid((D_IN_PROJ + 127) / 128, NTOK / 64);
      gemm_bf16f32<<<grid, 256, 0, stream>>>(Xn, WinT, zx, NTOK, D_IN_PROJ,
                                             D_MODEL, nullptr);
    }

    // causal dwconv + silu
    {
      size_t n = (size_t)NTOK * XBC_DIM;
      conv_silu_kernel<<<(unsigned)((n + 255) / 256), 256, 0, stream>>>(
          zx, pp.convw, pp.convb, xbc);
    }

    // dt / dA
    dt_kernel<<<(NTOK * NHEADS + 255) / 256, 256, 0, stream>>>(
        zx, pp.dtbias, pp.Alog, dtv, dAv);

    // sequential SSD scan, fused +D*x
    ssd_scan_kernel<<<BATCH * NHEADS, 64, 0, stream>>>(xbc, dtv, dAv, pp.Dp, yb);

    // gate + rmsnorm -> bf16
    gate_rmsnorm_kernel<<<NTOK, 256, 0, stream>>>(yb, zx, pp.normw, ynorm);

    // out_proj GEMM: (8192 x 1024) x (1024 x 512)
    {
      dim3 grid((D_MODEL + 127) / 128, NTOK / 64);
      gemm_bf16f32<<<grid, 256, 0, stream>>>(ynorm, WoutT, fs, NTOK, D_MODEL,
                                             D_INNER, nullptr);
    }

    if (br == 0) {
      // factual: slice query rows, project with outp_w + outp_b
      size_t n = (size_t)NTOKQ * D_MODEL;
      slice_q_to_bf16<<<(unsigned)((n + 255) / 256), 256, 0, stream>>>(fs, A3);
      dim3 grid((D_MODEL + 127) / 128, NTOKQ / 64);
      gemm_bf16f32<<<grid, 256, 0, stream>>>(A3, outpT, out_fact, NTOKQ,
                                             D_MODEL, D_MODEL, outp_b);
    } else {
      // counterfactual: raw query-slice output
      size_t n = (size_t)NTOKQ * D_MODEL;
      slice_q_copy<<<(unsigned)((n + 255) / 256), 256, 0, stream>>>(fs, out_cf);
    }
  }
}
